// ScanCore_16784732193282
// MI455X (gfx1250) — matrix operand order repaired and checked by reading, not compiled, not run on hardware
//
#include <hip/hip_runtime.h>
#include <hip/hip_bf16.h>

typedef __attribute__((ext_vector_type(16))) _Float16 v16h;
typedef __attribute__((ext_vector_type(8)))  float    v8f;
typedef __attribute__((ext_vector_type(4)))  _Float16 h4;

namespace {
constexpr int B  = 4, T = 4096, D = 1024, H = 1024;
constexpr int N2 = 2 * H;            // 2048 output cols of each projection
constexpr int M  = B * T;            // 16384 rows
constexpr int BK = 32;               // WMMA K (f16)
constexpr int NK = D / BK;           // 32 K-steps
constexpr int BM = 128, BN = 64;     // block tile (per gate/value half)
constexpr int LDA = 40;              // padded f16 LDS row stride (80B, 16B-aligned)
constexpr int TC = 256, NC = T / TC; // scan chunking: 16 chunks of 256
}

// Plain PODs so __builtin_bit_cast is legal (HIP float4 has ctors).
struct __attribute__((aligned(16))) f4v { float x, y, z, w; };
struct F4x2 { f4v a, b; };

// ---------------------------------------------------------------------------
// Fused GEMM + bias + sigmoid gating.
//   A   : [M, D]   f32 row-major (x or previous layer h)
//   W   : [D, 2H]  f32 row-major
//   bias: [2H]
// Writes g = sigmoid(A@W[:, :H] + b[:H])           -> gbuf [M, H]
//        u = (1-g) * (A@W[:, H:] + b[H:])          -> ubuf [M, H]
// Block = 8 waves; tile 128x64 for BOTH the gate and the paired value columns.
// ---------------------------------------------------------------------------
__global__ __launch_bounds__(256)
void fused_gemm_gate(const float* __restrict__ A,
                     const float* __restrict__ W,
                     const float* __restrict__ bias,
                     float* __restrict__ gbuf,
                     float* __restrict__ ubuf) {
  __shared__ __attribute__((aligned(16))) _Float16 As[2][BM][LDA];
  __shared__ __attribute__((aligned(16))) _Float16 Bs[2][2][BN][LDA]; // [buf][g/v][n][k]

  const int tid  = threadIdx.x;
  const int lane = tid & 31;
  const int wave = tid >> 5;
  const int wm   = (wave & 3) * 32;   // wave M offset inside block tile (4 waves along M)
  const int wn   = (wave >> 2) * 32;  // wave N offset (2 waves along N)
  const int half = lane >> 4;         // half-wave select per ISA layouts
  const int l16  = lane & 15;

  const int m_blk = blockIdx.x * BM;
  const int n_blk = blockIdx.y * BN;

  v8f accg[2][2] = {};
  v8f accv[2][2] = {};

  f4v ra[4];       // staged A: 128x32 f32 / 256 thr = 4 float4 each
  f4v rb[2][2];    // staged W gate/value: 32x64 f32 each / 256 thr = 2 float4 each

  auto gload = [&](int k0) {
#pragma unroll
    for (int i = 0; i < 4; ++i) {
      int id = tid + i * 256;          // 0..1023
      int r  = id >> 3;                // row 0..127
      int kq = id & 7;                 // float4 within the 32 k's
      ra[i] = *reinterpret_cast<const f4v*>(&A[(size_t)(m_blk + r) * D + k0 + kq * 4]);
    }
#pragma unroll
    for (int hlf = 0; hlf < 2; ++hlf) {
#pragma unroll
      for (int i = 0; i < 2; ++i) {
        int id = tid + i * 256;        // 0..511
        int k  = id >> 4;              // 0..31
        int nq = id & 15;              // float4 within the 64 n's
        int ncol = n_blk + nq * 4 + hlf * H;
        rb[hlf][i] = *reinterpret_cast<const f4v*>(&W[(size_t)(k0 + k) * N2 + ncol]);
      }
    }
  };

  auto sstore = [&](int s) {
#pragma unroll
    for (int i = 0; i < 4; ++i) {
      int id = tid + i * 256;
      int r  = id >> 3;
      int kq = id & 7;
      h4 hv = { (_Float16)ra[i].x, (_Float16)ra[i].y, (_Float16)ra[i].z, (_Float16)ra[i].w };
      *reinterpret_cast<h4*>(&As[s][r][kq * 4]) = hv;   // 8B aligned ds_store_b64
    }
#pragma unroll
    for (int hlf = 0; hlf < 2; ++hlf) {
#pragma unroll
      for (int i = 0; i < 2; ++i) {
        int id = tid + i * 256;
        int k  = id >> 4;
        int nq = (id & 15) * 4;
        Bs[s][hlf][nq + 0][k] = (_Float16)rb[hlf][i].x; // transpose into [n][k]
        Bs[s][hlf][nq + 1][k] = (_Float16)rb[hlf][i].y;
        Bs[s][hlf][nq + 2][k] = (_Float16)rb[hlf][i].z;
        Bs[s][hlf][nq + 3][k] = (_Float16)rb[hlf][i].w;
      }
    }
  };

  gload(0);
  sstore(0);

  for (int ks = 0; ks < NK; ++ks) {
    __syncthreads();                       // one barrier/iter (double buffered)
    const int s = ks & 1;
    const bool nxt = (ks + 1) < NK;
    if (nxt) gload((ks + 1) * BK);

    // --- gather WMMA fragments from LDS (contiguous-k order k = 16*half + s, b128 reads) ---
    v16h afr[2], bgfr[2], bvfr[2];
#pragma unroll
    for (int tm = 0; tm < 2; ++tm) {
      // A 16x32 f16: lanes 0-15 -> K 0..15, lanes 16-31 -> K 16..31
      const _Float16* p = &As[s][wm + tm * 16 + l16][half * 16];
      F4x2 t;
      t.a = *reinterpret_cast<const f4v*>(p);
      t.b = *reinterpret_cast<const f4v*>(p + 8);
      afr[tm] = __builtin_bit_cast(v16h, t);
    }
#pragma unroll
    for (int tn = 0; tn < 2; ++tn) {
      // B 32x16 f16: lane = column; lanes 0-15 K=0..15, lanes 16-31 K=16..31
      const _Float16* pg = &Bs[s][0][wn + tn * 16 + l16][half * 16];
      F4x2 tg;
      tg.a = *reinterpret_cast<const f4v*>(pg);
      tg.b = *reinterpret_cast<const f4v*>(pg + 8);
      bgfr[tn] = __builtin_bit_cast(v16h, tg);
      const _Float16* pv = &Bs[s][1][wn + tn * 16 + l16][half * 16];
      F4x2 tv;
      tv.a = *reinterpret_cast<const f4v*>(pv);
      tv.b = *reinterpret_cast<const f4v*>(pv + 8);
      bvfr[tn] = __builtin_bit_cast(v16h, tv);
    }

#pragma unroll
    for (int tm = 0; tm < 2; ++tm)
#pragma unroll
      for (int tn = 0; tn < 2; ++tn) {
        accg[tm][tn] = __builtin_amdgcn_wmma_f32_16x16x32_f16(
            false, afr[tm], false, bgfr[tn], (short)0, accg[tm][tn], false, false);
        accv[tm][tn] = __builtin_amdgcn_wmma_f32_16x16x32_f16(
            false, afr[tm], false, bvfr[tn], (short)0, accv[tm][tn], false, false);
      }

    if (nxt) sstore(s ^ 1);
  }

  // --- fused epilogue: bias + sigmoid + (1-g)*v ---
  // C layout: VGPR r -> (M = r + 8*half, N = lane&15)
#pragma unroll
  for (int tn = 0; tn < 2; ++tn) {
    const int gn = n_blk + wn + tn * 16 + l16;
    const float bg = bias[gn];
    const float bv = bias[H + gn];
#pragma unroll
    for (int tm = 0; tm < 2; ++tm) {
#pragma unroll
      for (int i = 0; i < 8; ++i) {
        const int gm = m_blk + wm + tm * 16 + half * 8 + i;
        float pg = accg[tm][tn][i] + bg;
        float pv = accv[tm][tn][i] + bv;
        float gs = 1.0f / (1.0f + __expf(-pg));
        gbuf[(size_t)gm * H + gn] = gs;
        ubuf[(size_t)gm * H + gn] = (1.0f - gs) * pv;
      }
    }
  }
}

// ---------------------------------------------------------------------------
// Two-level parallel scan of h_t = g_t * h_{t-1} + u_t along T.
// ---------------------------------------------------------------------------

// Pass 1: per (b, chunk, channel) fold (A,U) composition of the chunk.
__global__ __launch_bounds__(256)
void scan_chunk_reduce(const float* __restrict__ g, const float* __restrict__ u,
                       float* __restrict__ Ac, float* __restrict__ Uc) {
  const int j = blockIdx.x * 256 + threadIdx.x;
  const int c = blockIdx.y;
  const int b = blockIdx.z;
  size_t base = ((size_t)b * T + (size_t)c * TC) * H + j;
  float Aa = 1.0f, Uu = 0.0f;
  for (int t = 0; t < TC; ++t) {
    float a = g[base + (size_t)t * H];
    float w = u[base + (size_t)t * H];
    Uu = Uu * a + w;   // (A,U) o (a,w) = (A*a, U*a + w)
    Aa *= a;
  }
  size_t o = ((size_t)b * NC + c) * H + j;
  Ac[o] = Aa;
  Uc[o] = Uu;
}

// Pass 2: sequential over the NC=16 chunk aggregates; h0 is known, so carry h.
__global__ __launch_bounds__(256)
void scan_chunk_prefix(const float* __restrict__ Ac, const float* __restrict__ Uc,
                       const float* __restrict__ h0, float* __restrict__ Hstart,
                       float* __restrict__ state_out) {
  const int idx = blockIdx.x * 256 + threadIdx.x;  // b*H + j
  const int b = idx / H, j = idx % H;
  float h = h0[idx];
  for (int c = 0; c < NC; ++c) {
    size_t o = ((size_t)b * NC + c) * H + j;
    Hstart[o] = h;
    h = Ac[o] * h + Uc[o];
  }
  state_out[idx] = h;   // final per-channel state
}

// Pass 3: replay each chunk from its known start value, emit all h_t.
__global__ __launch_bounds__(256)
void scan_chunk_apply(const float* __restrict__ g, const float* __restrict__ u,
                      const float* __restrict__ Hstart, float* __restrict__ hout) {
  const int j = blockIdx.x * 256 + threadIdx.x;
  const int c = blockIdx.y;
  const int b = blockIdx.z;
  float h = Hstart[((size_t)b * NC + c) * H + j];
  size_t base = ((size_t)b * T + (size_t)c * TC) * H + j;
  for (int t = 0; t < TC; ++t) {
    h = g[base + (size_t)t * H] * h + u[base + (size_t)t * H];
    hout[base + (size_t)t * H] = h;
  }
}

// ---------------------------------------------------------------------------
extern "C" void kernel_launch(void* const* d_in, const int* in_sizes, int n_in,
                              void* d_out, int out_size, void* d_ws, size_t ws_size,
                              hipStream_t stream) {
  const float* x  = (const float*)d_in[0];
  const float* W0 = (const float*)d_in[1];
  const float* b0 = (const float*)d_in[2];
  const float* W1 = (const float*)d_in[3];
  const float* b1 = (const float*)d_in[4];
  const float* st = (const float*)d_in[5];   // [L, B, H]

  float* out_h  = (float*)d_out;                       // [B, T, H]
  float* out_st = out_h + (size_t)B * T * H;           // [L, B, H]

  // Workspace layout (~202 MB): g, u, inter-layer h (64 MB each) + scan temps.
  float* g  = (float*)d_ws;
  float* u  = g  + (size_t)M * H;
  float* h1 = u  + (size_t)M * H;
  float* Ac = h1 + (size_t)M * H;
  float* Uc = Ac + (size_t)B * NC * H;
  float* Hs = Uc + (size_t)B * NC * H;

  dim3 gblk(M / BM, H / BN);        // 128 x 16 blocks
  dim3 sgrid(H / 256, NC, B);       // 4 x 16 x 4
  const int pgrid = (B * H) / 256;  // 16

  // ---- layer 0 ----
  fused_gemm_gate<<<gblk, 256, 0, stream>>>(x, W0, b0, g, u);
  scan_chunk_reduce<<<sgrid, 256, 0, stream>>>(g, u, Ac, Uc);
  scan_chunk_prefix<<<pgrid, 256, 0, stream>>>(Ac, Uc, st, Hs, out_st);
  scan_chunk_apply<<<sgrid, 256, 0, stream>>>(g, u, Hs, h1);

  // ---- layer 1 ----
  fused_gemm_gate<<<gblk, 256, 0, stream>>>(h1, W1, b1, g, u);
  scan_chunk_reduce<<<sgrid, 256, 0, stream>>>(g, u, Ac, Uc);
  scan_chunk_prefix<<<pgrid, 256, 0, stream>>>(Ac, Uc, st + (size_t)B * H, Hs,
                                               out_st + (size_t)B * H);
  scan_chunk_apply<<<sgrid, 256, 0, stream>>>(g, u, Hs, out_h);
}
